// Codebook3D_20263655703265
// MI455X (gfx1250) — compile-verified, matrix operand order checked by reading, not run on hardware
//
#include <hip/hip_runtime.h>

// ---------------- problem constants (from reference) ----------------
#define K_CODES 8192
#define CDIM    256
#define NVEC    32768            // 4 * 8 * 32 * 32
#define DHW     8192             // 8 * 32 * 32
#define DECAYF  0.99f
#define OMDF    0.01f            // 1 - DECAY
#define BETAF   0.25f
#define EPSF    1e-5f
#define NTILES  (K_CODES / 16)   // 512 column tiles of 16 codes

typedef __attribute__((ext_vector_type(16))) __bf16 v16bf;
typedef __attribute__((ext_vector_type(8)))  __bf16 v8bf;
typedef __attribute__((ext_vector_type(8)))  float  v8f;

// ---------------------------------------------------------------
// 1) Initialize outputs: new_embed_avg = DECAY*ema_embed_avg,
//    new_cluster_size = DECAY*ema_cluster_size, loss = 0.
// ---------------------------------------------------------------
__global__ __launch_bounds__(256) void init_outputs(
    const float* __restrict__ ema_cs, const float* __restrict__ ema_avg,
    float* __restrict__ ncs, float* __restrict__ navg, float* __restrict__ loss)
{
    int i = blockIdx.x * 256 + threadIdx.x;
    navg[i] = DECAYF * ema_avg[i];
    if (i < K_CODES) ncs[i] = DECAYF * ema_cs[i];
    if (i == 0) *loss = 0.0f;
}

// ---------------------------------------------------------------
// 2) Convert codebook to bf16 + per-code squared norm of the
//    bf16-rounded values (consistent with the bf16 cross term).
// ---------------------------------------------------------------
__global__ __launch_bounds__(256) void convert_embed(
    const float* __restrict__ emb, __bf16* __restrict__ ebf,
    float* __restrict__ enorm)
{
    int k = blockIdx.x, c = threadIdx.x;
    float v = emb[(size_t)k * CDIM + c];
    __bf16 bv = (__bf16)v;
    ebf[(size_t)k * CDIM + c] = bv;
    float q = (float)bv;
    __shared__ float red[256];
    red[c] = q * q;
    __syncthreads();
    for (int o = 128; o > 0; o >>= 1) {
        if (c < o) red[c] += red[c + o];
        __syncthreads();
    }
    if (c == 0) enorm[k] = red[0];
}

// ---------------------------------------------------------------
// 3) Transpose z [B,C,D,H,W] -> z_flat [N,C] in bf16 via LDS.
// ---------------------------------------------------------------
__global__ __launch_bounds__(256) void transpose_z(
    const float* __restrict__ z, __bf16* __restrict__ zbf)
{
    __shared__ float lds[32][257];
    const int tid = threadIdx.x;
    const int nbase = blockIdx.x * 32;
    const int b  = nbase / DHW;
    const int s0 = nbase % DHW;

#pragma unroll 4
    for (int it = 0; it < 32; ++it) {
        int e  = it * 256 + tid;
        int c  = e >> 5;
        int sl = e & 31;
        lds[sl][c] = z[((size_t)b * CDIM + c) * DHW + (s0 + sl)];
    }
    __syncthreads();
#pragma unroll 4
    for (int it = 0; it < 32; ++it) {
        int e  = it * 256 + tid;
        int sl = e >> 8;
        int c  = e & 255;
        zbf[(size_t)(nbase + sl) * CDIM + c] = (__bf16)lds[sl][c];
    }
}

// ---------------------------------------------------------------
// Async copy of one 8 KB codebook panel (16 codes x 256 bf16,
// contiguous in memory) into LDS. Issued by one full wave: lane l
// moves 16 B per instruction, 16 instructions cover 8 KB.
// Tracked on ASYNCcnt.
// ---------------------------------------------------------------
__device__ __forceinline__ void issue_panel_async(
    const __bf16* ebf, int tile, const __bf16* lds_dst, int lane)
{
    unsigned long long g = (unsigned long long)(uintptr_t)ebf
                         + (unsigned long long)tile * (16u * CDIM * 2u)
                         + (unsigned)lane * 16u;
    unsigned l = (unsigned)(uintptr_t)lds_dst + (unsigned)lane * 16u;
#pragma unroll
    for (int i = 0; i < 16; ++i) {
        asm volatile("global_load_async_to_lds_b128 %0, %1, off"
                     :: "v"(l), "v"(g) : "memory");
        g += 512ull;
        l += 512u;
    }
}

// B fragment (16-bit WMMA operand) from an LDS panel row.
__device__ __forceinline__ v16bf load_bfrag(const __bf16* bp, int ks, int half)
{
    int klo = ks * 32 + half * 8;
    v8bf lo = *(const v8bf*)(bp + klo);
    v8bf hi = *(const v8bf*)(bp + klo + 16);
    return __builtin_shufflevector(lo, hi,
        0,1,2,3,4,5,6,7,8,9,10,11,12,13,14,15);
}

// ---------------------------------------------------------------
// 4) Fused distance + argmin GEMM on bf16 WMMA.
//    argmin_n(||z||^2 + ||e_n||^2 - 2 z.e_n) == argmin_n(||e_n||^2 - 2 z.e_n)
//    Block = 8 waves; each wave owns 32 z-rows (two A sets, two
//    accumulators -> each LDS B fragment feeds 2 WMMAs).
//    Codebook panels are double-buffered in LDS via async loads;
//    B fragments are double-buffered in VGPRs so the ds_loads for
//    step ks+1 are in flight while the WMMAs of step ks execute.
// ---------------------------------------------------------------
__global__ __launch_bounds__(256) void argmin_wmma(
    const __bf16* __restrict__ zbf, const __bf16* __restrict__ ebf,
    const float* __restrict__ enorm,
    int* __restrict__ inds, float* __restrict__ indsf)
{
    __shared__ __align__(16) __bf16 panel[2][16 * CDIM];   // 2 x 8 KB

    const int lane = threadIdx.x & 31;
    const int wave = threadIdx.x >> 5;
    const int m0   = (blockIdx.x * 8 + wave) * 32;
    const int r    = lane & 15;
    const int half = lane >> 4;

    // Preload A fragments: rows m0..m0+15 and m0+16..m0+31, full C=256.
    v16bf afrag0[8], afrag1[8];
    {
        const __bf16* a0 = zbf + (size_t)(m0 + r) * CDIM;
        const __bf16* a1 = zbf + (size_t)(m0 + 16 + r) * CDIM;
#pragma unroll
        for (int ks = 0; ks < 8; ++ks) {
            int klo = ks * 32 + half * 8;
            v8bf lo0 = *(const v8bf*)(a0 + klo);
            v8bf hi0 = *(const v8bf*)(a0 + klo + 16);
            afrag0[ks] = __builtin_shufflevector(lo0, hi0,
                0,1,2,3,4,5,6,7,8,9,10,11,12,13,14,15);
            v8bf lo1 = *(const v8bf*)(a1 + klo);
            v8bf hi1 = *(const v8bf*)(a1 + klo + 16);
            afrag1[ks] = __builtin_shufflevector(lo1, hi1,
                0,1,2,3,4,5,6,7,8,9,10,11,12,13,14,15);
        }
    }

    float best0[8], best1[8];
    int   bidx0[8], bidx1[8];
#pragma unroll
    for (int j = 0; j < 8; ++j) {
        best0[j] = 3.4e38f; bidx0[j] = 0;
        best1[j] = 3.4e38f; bidx1[j] = 0;
    }

    // Prologue: stage tile 0 into buffer 0.
    if (wave == 0) {
        issue_panel_async(ebf, 0, &panel[0][0], lane);
        asm volatile("s_wait_asynccnt 0x0" ::: "memory");
    }
    __syncthreads();

    for (int t = 0; t < NTILES; ++t) {
        const int p  = t & 1;
        const int nb = t * 16;

        // Kick off next panel into the other buffer.
        if (wave == 0 && (t + 1) < NTILES)
            issue_panel_async(ebf, t + 1, &panel[1 - p][0], lane);

        const __bf16* bp = &panel[p][r * CDIM];
        const float en = enorm[nb + r];         // overlaps with LDS traffic
        const int   nc = nb + r;

        // Software-pipelined k-loop: loads for ks+1 fly over WMMAs of ks.
        v8f acc0 = {}, acc1 = {};
        v16bf bcur = load_bfrag(bp, 0, half);
#pragma unroll
        for (int ks = 0; ks < 8; ++ks) {
            v16bf bnxt = bcur;
            if (ks < 7) bnxt = load_bfrag(bp, ks + 1, half);
            acc0 = __builtin_amdgcn_wmma_f32_16x16x32_bf16(
                false, afrag0[ks], false, bcur, (short)0, acc0, false, false);
            acc1 = __builtin_amdgcn_wmma_f32_16x16x32_bf16(
                false, afrag1[ks], false, bcur, (short)0, acc1, false, false);
            bcur = bnxt;
        }

#pragma unroll
        for (int j = 0; j < 8; ++j) {
            float d0 = __builtin_fmaf(acc0[j], -2.0f, en);
            if (d0 < best0[j]) { best0[j] = d0; bidx0[j] = nc; }
            float d1 = __builtin_fmaf(acc1[j], -2.0f, en);
            if (d1 < best1[j]) { best1[j] = d1; bidx1[j] = nc; }
        }

        // Publish next buffer / protect current one.
        if (wave == 0 && (t + 1) < NTILES)
            asm volatile("s_wait_asynccnt 0x0" ::: "memory");
        __syncthreads();
    }

    // Min-reduce across the 16 lanes of each half (xor masks < 16 stay in-half).
#pragma unroll
    for (int m = 8; m >= 1; m >>= 1) {
#pragma unroll
        for (int j = 0; j < 8; ++j) {
            float ov0 = __shfl_xor(best0[j], m, 32);
            int   oi0 = __shfl_xor(bidx0[j], m, 32);
            if (ov0 < best0[j] || (ov0 == best0[j] && oi0 < bidx0[j])) {
                best0[j] = ov0; bidx0[j] = oi0;
            }
            float ov1 = __shfl_xor(best1[j], m, 32);
            int   oi1 = __shfl_xor(bidx1[j], m, 32);
            if (ov1 < best1[j] || (ov1 == best1[j] && oi1 < bidx1[j])) {
                best1[j] = ov1; bidx1[j] = oi1;
            }
        }
    }
    if (r == 0) {
#pragma unroll
        for (int j = 0; j < 8; ++j) {
            int row0 = m0 + j + half * 8;
            inds[row0]  = bidx0[j];
            indsf[row0] = (float)bidx0[j];
            int row1 = m0 + 16 + j + half * 8;
            inds[row1]  = bidx1[j];
            indsf[row1] = (float)bidx1[j];
        }
    }
}

// ---------------------------------------------------------------
// 5) Fused gather (z_q), commit loss, and EMA scatter.
// ---------------------------------------------------------------
__global__ __launch_bounds__(256) void gather_scatter(
    const float* __restrict__ z, const float* __restrict__ emb,
    const int* __restrict__ inds,
    float* __restrict__ zq, float* __restrict__ loss,
    float* __restrict__ ncs, float* __restrict__ navg)
{
    const int n = blockIdx.x;
    const int c = threadIdx.x;
    const int b = n / DHW;
    const int s = n % DHW;
    const int idx = inds[n];

    const size_t zoff = ((size_t)b * CDIM + c) * DHW + s;
    const float zv = z[zoff];
    const float ev = emb[(size_t)idx * CDIM + c];

    zq[zoff] = ev;
    atomicAdd(&navg[(size_t)idx * CDIM + c], OMDF * zv);
    if (c == 0) atomicAdd(&ncs[idx], OMDF);

    float d = ev - zv;
    __shared__ float red[256];
    red[c] = d * d;
    __syncthreads();
    for (int o = 128; o > 0; o >>= 1) {
        if (c < o) red[c] += red[c + o];
        __syncthreads();
    }
    if (c == 0) atomicAdd(loss, red[0] * (BETAF / 8388608.0f));
}

// ---------------------------------------------------------------
// 6) n = sum(new_cluster_size), single block.
// ---------------------------------------------------------------
__global__ __launch_bounds__(256) void reduce_nsum(
    const float* __restrict__ ncs, float* __restrict__ nsum)
{
    __shared__ float red[256];
    float s = 0.0f;
    for (int i = threadIdx.x; i < K_CODES; i += 256) s += ncs[i];
    red[threadIdx.x] = s;
    __syncthreads();
    for (int o = 128; o > 0; o >>= 1) {
        if (threadIdx.x < o) red[threadIdx.x] += red[threadIdx.x + o];
        __syncthreads();
    }
    if (threadIdx.x == 0) *nsum = red[0];
}

// ---------------------------------------------------------------
// 7) new_embedding = new_embed_avg / max(smoothed, 1e-6)
// ---------------------------------------------------------------
__global__ __launch_bounds__(256) void finalize_embedding(
    const float* __restrict__ navg, const float* __restrict__ ncs,
    const float* __restrict__ nsum_p, float* __restrict__ nemb)
{
    const int k = blockIdx.x, c = threadIdx.x;
    const float n  = *nsum_p;
    const float sm = (ncs[k] + EPSF) / (n + (float)K_CODES * EPSF) * n;
    const float dn = fmaxf(sm, 1e-6f);
    nemb[(size_t)k * CDIM + c] = navg[(size_t)k * CDIM + c] / dn;
}

// ---------------------------------------------------------------
extern "C" void kernel_launch(void* const* d_in, const int* in_sizes, int n_in,
                              void* d_out, int out_size, void* d_ws, size_t ws_size,
                              hipStream_t stream)
{
    (void)in_sizes; (void)n_in; (void)out_size; (void)ws_size;

    const float* z       = (const float*)d_in[0];   // [4,256,8,32,32]
    const float* emb     = (const float*)d_in[1];   // [8192,256]
    const float* ema_cs  = (const float*)d_in[2];   // [8192]
    const float* ema_avg = (const float*)d_in[3];   // [8192,256]

    float* out      = (float*)d_out;
    float* zq_out   = out;                 // 8388608
    float* inds_f   = out + 8388608;       //   32768
    float* loss_out = out + 8421376;       //       1
    float* new_emb  = out + 8421377;       // 2097152
    float* ncs      = out + 10518529;      //    8192
    float* navg     = out + 10526721;      // 2097152

    char* w = (char*)d_ws;
    __bf16* z_bf  = (__bf16*)w;  w += (size_t)NVEC   * CDIM * sizeof(__bf16);
    __bf16* e_bf  = (__bf16*)w;  w += (size_t)K_CODES* CDIM * sizeof(__bf16);
    float*  enorm = (float*)w;   w += (size_t)K_CODES * sizeof(float);
    int*    inds  = (int*)w;     w += (size_t)NVEC    * sizeof(int);
    float*  nsum  = (float*)w;

    init_outputs      <<<dim3((K_CODES * CDIM) / 256), dim3(256), 0, stream>>>(ema_cs, ema_avg, ncs, navg, loss_out);
    convert_embed     <<<dim3(K_CODES),                dim3(256), 0, stream>>>(emb, e_bf, enorm);
    transpose_z       <<<dim3(NVEC / 32),              dim3(256), 0, stream>>>(z, z_bf);
    argmin_wmma       <<<dim3(NVEC / 256),             dim3(256), 0, stream>>>(z_bf, e_bf, enorm, inds, inds_f);
    gather_scatter    <<<dim3(NVEC),                   dim3(256), 0, stream>>>(z, emb, inds, zq_out, loss_out, ncs, navg);
    reduce_nsum       <<<dim3(1),                      dim3(256), 0, stream>>>(ncs, nsum);
    finalize_embedding<<<dim3(K_CODES),                dim3(256), 0, stream>>>(navg, ncs, nsum, new_emb);
}
